// LSTMEncoder_5583457484844
// MI455X (gfx1250) — matrix operand order repaired and checked by reading, not compiled, not run on hardware
//
#include <hip/hip_runtime.h>
#include <cstdint>

typedef __attribute__((ext_vector_type(16))) _Float16 v16h;
typedef __attribute__((ext_vector_type(8)))  float    v8f;
typedef __attribute__((ext_vector_type(2)))  __fp16   v2fp16;

#define T_STEPS 140
#define HID 10
#define GATES 40
#define WAVES_PER_BLOCK 4
#define ROWS_PER_WAVE 16   // batch rows (N columns of WMMA) per wave
#define X_PER_WAVE (ROWS_PER_WAVE * T_STEPS)   // 2240 floats, contiguous in global

// ---- Tensor Data Mover availability / arity detection -----------------------
#if defined(__has_builtin)
#  if __has_builtin(__builtin_amdgcn_tensor_load_to_lds)
#    define HAVE_TDM 1
#  endif
#endif
#if __has_include(<hip/amd_detail/amd_gfx1250_TDM.h>)
#  define TDM_ARITY6 1     // therock/clang-23 headers -> 6-arg builtin
#else
#  define TDM_ARITY6 0     // ROCm 7.2 / clang-22 -> 5-arg builtin
#endif

#ifdef HAVE_TDM
typedef __attribute__((ext_vector_type(4))) unsigned int v4u;
typedef __attribute__((ext_vector_type(8))) int          v8i_t;
typedef __attribute__((ext_vector_type(4))) int          v4i_t;

// One 1D TDM descriptor: copy nElems f32 from gsrc -> LDS[lds_byte_off].
__device__ __forceinline__ void tdm_load_x(const float* gsrc, unsigned lds_byte_off,
                                           unsigned nElems) {
    const unsigned long long ga = (unsigned long long)(uintptr_t)gsrc;
    v4u g0;
    g0[0] = 1u;                                             // count=1, user desc
    g0[1] = lds_byte_off;                                   // lds_addr (bytes)
    g0[2] = (unsigned)(ga & 0xFFFFFFFFu);                   // global_addr[31:0]
    g0[3] = (unsigned)((ga >> 32) & 0x01FFFFFFu)            // global_addr[56:32]
          | (2u << 30);                                     // type=2 ("image")
    v8i_t g1;
    g1[0] = (int)(2u << 16);                                // data_size=2 -> 4B elems
    g1[1] = (int)((nElems & 0xFFFFu) << 16);                // tensor_dim0[15:0] @ bit48
    g1[2] = (int)((nElems >> 16) & 0xFFFFu) | (1 << 16);    // tensor_dim0[31:16], tensor_dim1=1
    g1[3] = (int)((nElems & 0xFFFFu) << 16);                // tile_dim0 = nElems
    g1[4] = 1;                                              // tile_dim1=1, tile_dim2=0
    g1[5] = (int)nElems;                                    // tensor_dim0_stride[31:0]
    g1[6] = 0;
    g1[7] = 0;
    const v4i_t gz = {0, 0, 0, 0};
#if TDM_ARITY6
    const v8i_t gz8 = {0, 0, 0, 0, 0, 0, 0, 0};
    __builtin_amdgcn_tensor_load_to_lds(g0, g1, gz, gz, gz8, 0);
#else
    __builtin_amdgcn_tensor_load_to_lds(g0, g1, gz, gz, 0);
#endif
}
#endif // HAVE_TDM

// sigmoid(x) = 1 / (1 + 2^(-x*log2 e))  -> v_exp_f32 + v_rcp_f32
__device__ __forceinline__ float sigmoid_fast(float x) {
    float e = __builtin_amdgcn_exp2f(-1.44269504088896341f * x);
    return __builtin_amdgcn_rcpf(1.0f + e);
}

// pack two f32 -> packed f16 pair (v_cvt_pk_rtz_f16_f32), as raw u32
__device__ __forceinline__ unsigned pk(float lo, float hi) {
    union { v2fp16 h; unsigned u; } c;
    c.h = __builtin_amdgcn_cvt_pkrtz(lo, hi);
    return c.u;
}

union AV { unsigned u[8]; v16h v; };

__global__ __launch_bounds__(WAVES_PER_BLOCK * 32)
void lstm_wmma_kernel(const float* __restrict__ x,
                      const float* __restrict__ W,
                      const float* __restrict__ U,
                      const float* __restrict__ bias,
                      float* __restrict__ out)
{
    __shared__ float sx[WAVES_PER_BLOCK * X_PER_WAVE];

    const int lane  = threadIdx.x & 31;
    const int wave  = threadIdx.x >> 5;
    const int gwave = blockIdx.x * WAVES_PER_BLOCK + wave;
    const int m0    = gwave * ROWS_PER_WAVE;           // first batch row of this wave
    const int l15   = lane & 15;
    const bool lowHalf = lane < 16;

    // ---------------- stage this wave's x block (16 rows x 140 = one contiguous
    // 8960B region) to LDS.  TDM (tensor_load_to_lds) when available, else float4 copy.
#ifdef HAVE_TDM
    {
        // sx is the only static LDS object -> its allocation offset is 0.
        tdm_load_x(x + (size_t)m0 * T_STEPS,
                   (unsigned)(wave * X_PER_WAVE * sizeof(float)),
                   (unsigned)X_PER_WAVE);
        __builtin_amdgcn_s_wait_tensorcnt(0);
        // Escape sx into an opaque side-effecting op: the TDM intrinsic's LDS
        // writes are invisible to alias analysis, and without this the compiler
        // folds all sx loads to undef/zero.  With the address escaped + memory
        // clobber, the loads stay real and keep addrspace(3) (ds_load_b32).
        asm volatile("" : : "s"(sx) : "memory");
    }
#else
    {
        const float4* src = (const float4*)(x + (size_t)m0 * T_STEPS);
        float4*       dst = (float4*)(sx + wave * X_PER_WAVE);
        for (int i = lane; i < X_PER_WAVE / 4; i += 32)
            dst[i] = src[i];
    }
#endif
    __syncthreads();

    // ---------------- build loop-invariant A tiles: A[g][k] = U[k][g], A[g][10]=W[g], A[g][11]=b[g]
    // A in this code's contiguous k order (k = 16h + s): lanes 0-15 row M=l, K=0..11 in VGPR0-5 (pairs);
    //                     lanes 16-31 (K=16..31) zero.
    AV A[4];
    #pragma unroll
    for (int tau = 0; tau < 4; ++tau) {      // 0:i 1:f 2:g 3:o   col = tau*10 + unit
        float uc[10], wc = 0.f, bc = 0.f;
        #pragma unroll
        for (int k = 0; k < 10; ++k) uc[k] = 0.f;
        if (l15 < HID) {
            const int col = tau * HID + l15;
            #pragma unroll
            for (int k = 0; k < 10; ++k) uc[k] = U[k * GATES + col];
            wc = W[col];
            bc = bias[col];
        }
        const unsigned lo0 = pk(uc[0], uc[1]);
        const unsigned lo1 = pk(uc[2], uc[3]);
        const unsigned lo2 = pk(uc[4], uc[5]);
        const unsigned lo3 = pk(uc[6], uc[7]);
        const unsigned hi0 = pk(uc[8], uc[9]);
        const unsigned hi1 = pk(wc, bc);       // K=10 -> W, K=11 -> bias
        A[tau].u[0] = lowHalf ? lo0 : 0u;
        A[tau].u[1] = lowHalf ? lo1 : 0u;
        A[tau].u[2] = lowHalf ? lo2 : 0u;
        A[tau].u[3] = lowHalf ? lo3 : 0u;
        A[tau].u[4] = lowHalf ? hi0 : 0u; A[tau].u[5] = lowHalf ? hi1 : 0u; A[tau].u[6] = 0u; A[tau].u[7] = 0u;
    }

    // ---------------- recurrent state (per lane: batch column n = l15)
    // lanes 0-15: units 0..7 of batch n; lanes 16-31: units 8..9 of batch n (slots 0,1)
    float c[8];
    float hcur[8];
    #pragma unroll
    for (int r = 0; r < 8; ++r) { c[r] = 0.f; hcur[r] = 0.f; }
    unsigned h01 = 0u, h23 = 0u, h45 = 0u, h67 = 0u;

    const float* xrow = sx + wave * X_PER_WAVE + l15 * T_STEPS;
    const int bperm_idx = (l15 + 16) << 2;   // fetch packed h[8],h[9] from upper lane half

    for (int t = 0; t < T_STEPS; ++t) {
        const float xv = xrow[t];
        const unsigned pkx = pk(xv, 1.0f);   // K=10 -> x_t, K=11 -> 1.0 (bias row)
        // upper-lane h01 holds pack(h[n][8], h[n][9])
        const unsigned h89 = (unsigned)__builtin_amdgcn_ds_bpermute(bperm_idx, (int)h01);

        // B in the same contiguous k order as A: lanes 0-15 carry K=0..15 (packed pairs per VGPR);
        // lanes 16-31 carry K=16..31.  A is identically zero for K>=12, so the
        // upper-lane contents (always finite) contribute 0 -- no masking needed.
        union { unsigned u[8]; v16h v; } Bm;
        Bm.u[0] = h01;
        Bm.u[1] = h23;
        Bm.u[2] = h45;
        Bm.u[3] = h67;
        Bm.u[4] = h89;
        Bm.u[5] = pkx;
        Bm.u[6] = 0u;
        Bm.u[7] = 0u;

        const v8f zacc = {};
        // z = A_tau x B : one 16x16 tile per gate type, C = 0 (bias folded into K=11)
        v8f zi = __builtin_amdgcn_wmma_f32_16x16x32_f16(false, A[0].v, false, Bm.v, (short)0, zacc, false, false);
        v8f zf = __builtin_amdgcn_wmma_f32_16x16x32_f16(false, A[1].v, false, Bm.v, (short)0, zacc, false, false);
        v8f zg = __builtin_amdgcn_wmma_f32_16x16x32_f16(false, A[2].v, false, Bm.v, (short)0, zacc, false, false);
        v8f zo = __builtin_amdgcn_wmma_f32_16x16x32_f16(false, A[3].v, false, Bm.v, (short)0, zacc, false, false);

        // lane-local gate math: slot r = unit r (lanes<16) / unit 8+r (lanes>=16)
        #pragma unroll
        for (int r = 0; r < 8; ++r) {
            const float si = sigmoid_fast(zi[r]);
            const float sf = sigmoid_fast(zf[r]);
            const float sg = sigmoid_fast(zg[r]);
            const float so = sigmoid_fast(zo[r]);
            c[r]    = sf * c[r] + si * sg;
            hcur[r] = so * sigmoid_fast(c[r]);
        }
        h01 = pk(hcur[0], hcur[1]);
        h23 = pk(hcur[2], hcur[3]);
        h45 = pk(hcur[4], hcur[5]);
        h67 = pk(hcur[6], hcur[7]);
    }

    // ---------------- write final h: out[(m0+n)*10 + j], f32
    const size_t rowBase = (size_t)(m0 + l15) * HID;
    if (lowHalf) {
        #pragma unroll
        for (int r = 0; r < 8; ++r) out[rowBase + r] = hcur[r];
    } else {
        out[rowBase + 8] = hcur[0];
        out[rowBase + 9] = hcur[1];
    }
}

extern "C" void kernel_launch(void* const* d_in, const int* in_sizes, int n_in,
                              void* d_out, int out_size, void* d_ws, size_t ws_size,
                              hipStream_t stream) {
    const float* x  = (const float*)d_in[0];
    const float* W  = (const float*)d_in[1];
    const float* U  = (const float*)d_in[2];
    const float* b  = (const float*)d_in[3];
    float* out = (float*)d_out;

    const int Bn = 131072;
    const int rowsPerBlock = ROWS_PER_WAVE * WAVES_PER_BLOCK;   // 64
    dim3 grid(Bn / rowsPerBlock);                               // 2048 blocks, 8192 waves
    dim3 block(WAVES_PER_BLOCK * 32);                           // 128 threads = 4 waves
    lstm_wmma_kernel<<<grid, block, 0, stream>>>(x, W, U, b, out);
}
